// ScheduleGNN_19894288515200
// MI455X (gfx1250) — matrix operand order repaired and checked by reading, not compiled, not run on hardware
//
#include <hip/hip_runtime.h>
#include <hip/hip_bf16.h>

// ---------------------------------------------------------------------------
// ScheduleGNN for MI455X (gfx1250, wave32).
// Heavy contractions (node GEMMs + 260x128 edge MLP) run on
// v_wmma_f32_16x16x32_bf16 with fp32 accumulation; softmax / scatter stay fp32.
// ---------------------------------------------------------------------------

#define NN   50000          // nodes
#define NE   800000         // edges
#define ESL  (NE + NN)      // edges + self loops
#define DD   3
#define EDIM 4
#define HH   128
#define NEG_SLOPE 0.2f

typedef __attribute__((ext_vector_type(16))) __bf16 v16bf;
typedef __attribute__((ext_vector_type(8)))  float  v8f;

union FragBF {                 // 16 bf16 = 8 dwords, per lane WMMA operand
    v16bf    v;
    unsigned d[8];
};

__device__ __forceinline__ unsigned short f2bf(float f) {
    unsigned u = __float_as_uint(f);
    unsigned r = u + 0x7FFFu + ((u >> 16) & 1u);   // round-to-nearest-even
    return (unsigned short)(r >> 16);
}

// monotone order-preserving float->uint key (for atomicMax segment-max)
__device__ __forceinline__ unsigned enc_f(float f) {
    unsigned u = __float_as_uint(f);
    return (u & 0x80000000u) ? ~u : (u | 0x80000000u);
}
__device__ __forceinline__ float dec_f(unsigned k) {
    unsigned u = (k & 0x80000000u) ? (k & 0x7FFFFFFFu) : ~k;
    return __uint_as_float(u);
}

// ---------------------------------------------------------------------------
// 0) mean(edge_attr) for PyG-style self loops
// ---------------------------------------------------------------------------
__global__ void k_ea_mean(const float* __restrict__ ea, float* __restrict__ out) {
    __shared__ float red[256][4];
    float s0 = 0.f, s1 = 0.f, s2 = 0.f, s3 = 0.f;
    for (int e = threadIdx.x; e < NE; e += 256) {
        float4 v = reinterpret_cast<const float4*>(ea)[e];
        s0 += v.x; s1 += v.y; s2 += v.z; s3 += v.w;
    }
    red[threadIdx.x][0] = s0; red[threadIdx.x][1] = s1;
    red[threadIdx.x][2] = s2; red[threadIdx.x][3] = s3;
    __syncthreads();
    for (int st = 128; st > 0; st >>= 1) {
        if (threadIdx.x < st)
            for (int d = 0; d < 4; ++d)
                red[threadIdx.x][d] += red[threadIdx.x + st][d];
        __syncthreads();
    }
    if (threadIdx.x < 4) out[threadIdx.x] = red[0][threadIdx.x] * (1.0f / (float)NE);
}

// ---------------------------------------------------------------------------
// 1) node encoders (D=3 -> H), masked by node index (job: n < N/2)
// ---------------------------------------------------------------------------
__global__ void k_encode(const float* __restrict__ x,
                         const float* __restrict__ Wj,  const float* __restrict__ bj,
                         const float* __restrict__ Wrt, const float* __restrict__ brt,
                         unsigned short* __restrict__ hb) {
    int idx = blockIdx.x * 256 + threadIdx.x;
    if (idx >= NN * HH) return;
    int n = idx >> 7, f = idx & 127;
    const float* W; const float* b;
    if (n < NN / 2) { W = Wj; b = bj; } else { W = Wrt; b = brt; }
    float x0 = x[n * 3 + 0], x1 = x[n * 3 + 1], x2 = x[n * 3 + 2];
    float v = b[f] + x0 * W[f] + x1 * W[HH + f] + x2 * W[2 * HH + f];
    hb[idx] = f2bf(fmaxf(v, 0.f));
}

// ---------------------------------------------------------------------------
// 2) pack fp32 weight [K x 128] into bf16 WMMA B-fragment layout.
// Layout consumed as: for (ntile, ktile) lane holds 16 contiguous bf16:
//   B[k = kt*32 + 16*(j>>3) + 8*(lane>>4) + (j&7)][n = nt*16 + (lane&15)]
// ---------------------------------------------------------------------------
__global__ void k_pack_b(const float* __restrict__ W, unsigned short* __restrict__ Bp,
                         int K, int KT) {
    int idx = blockIdx.x * 256 + threadIdx.x;
    int total = KT * 8 * 512;
    if (idx >= total) return;
    int j = idx & 15;
    int lane = (idx >> 4) & 31;
    int rem = idx >> 9;
    int kt = rem % KT;
    int k = kt * 32 + ((j & 8) << 1) + ((lane >> 4) << 3) + (j & 7);
    int n = (rem / KT) * 16 + (lane & 15);
    float v = (k < K) ? W[k * HH + n] : 0.f;
    Bp[idx] = f2bf(v);
}

// ---------------------------------------------------------------------------
// 3) per-layer init: segment-max keys, segment sums, output accumulator
// ---------------------------------------------------------------------------
__global__ void k_layer_init(unsigned* __restrict__ nmax, float* __restrict__ nsum,
                             float* __restrict__ accum) {
    int idx = blockIdx.x * 256 + threadIdx.x;
    if (idx < NN) { nmax[idx] = 0u; nsum[idx] = 0.f; }
    if (idx < NN * HH) accum[idx] = 0.f;
}

// ---------------------------------------------------------------------------
// 4) xl = h@Wl, xr = h@Wr fused (A fragments reused). One wave = 16x16 tile.
//    Block = 8 waves covering all 8 N-tiles of one 16-row M-tile.
// ---------------------------------------------------------------------------
__global__ void __launch_bounds__(256)
k_gemm_xlxr(const unsigned short* __restrict__ hb,
            const unsigned short* __restrict__ BpL,
            const unsigned short* __restrict__ BpR,
            float* __restrict__ xl, float* __restrict__ xr) {
    int lane = threadIdx.x & 31;
    int wv   = threadIdx.x >> 5;          // = N tile
    int kh   = lane >> 4;
    int row  = blockIdx.x * 16 + (lane & 15);
    const unsigned* hrow = reinterpret_cast<const unsigned*>(hb + (size_t)row * HH);

    v8f accL = {}; v8f accR = {};
#pragma unroll
    for (int kt = 0; kt < 4; ++kt) {
        FragBF a;
#pragma unroll
        for (int t = 0; t < 8; ++t) {
            int k = kt * 32 + ((t & 4) ? 16 : 0) + (t & 3) * 2 + kh * 8;
            a.d[t] = hrow[k >> 1];
        }
        const unsigned* bl = reinterpret_cast<const unsigned*>(
            BpL + (((size_t)wv * 4 + kt) * 32 + lane) * 16);
        const unsigned* br = reinterpret_cast<const unsigned*>(
            BpR + (((size_t)wv * 4 + kt) * 32 + lane) * 16);
        FragBF b1, b2;
#pragma unroll
        for (int t = 0; t < 8; ++t) { b1.d[t] = bl[t]; b2.d[t] = br[t]; }
        accL = __builtin_amdgcn_wmma_f32_16x16x32_bf16(false, a.v, false, b1.v,
                                                       (short)0, accL, false, false);
        accR = __builtin_amdgcn_wmma_f32_16x16x32_bf16(false, a.v, false, b2.v,
                                                       (short)0, accR, false, false);
    }
    int col = wv * 16 + (lane & 15);
#pragma unroll
    for (int r = 0; r < 8; ++r) {
        int m = blockIdx.x * 16 + r + kh * 8;
        xl[(size_t)m * HH + col] = accL[r];
        xr[(size_t)m * HH + col] = accR[r];
    }
}

// ---------------------------------------------------------------------------
// 5) GATv2 score pass: one wave per edge, 4 features/lane.
//    score = dot(leaky_relu(xl[src]+xr[dst]+ea@We), att); atomicMax into dst.
// ---------------------------------------------------------------------------
__global__ void __launch_bounds__(256)
k_edge_score(const float* __restrict__ xl, const float* __restrict__ xr,
             const float* __restrict__ ea, const int* __restrict__ ei,
             const float* __restrict__ We, const float* __restrict__ att,
             const float* __restrict__ eam,
             float* __restrict__ sbuf, unsigned* __restrict__ nmax) {
    __shared__ float sWe[EDIM * HH];
    __shared__ float sAtt[HH];
    for (int i = threadIdx.x; i < EDIM * HH; i += 256) sWe[i] = We[i];
    if (threadIdx.x < HH) sAtt[threadIdx.x] = att[threadIdx.x];
    __syncthreads();

    int e = (blockIdx.x * 256 + threadIdx.x) >> 5;
    int lane = threadIdx.x & 31;
    if (e >= ESL) return;

    int src, dst; float e0, e1, e2, e3;
    if (e < NE) {
        src = ei[e]; dst = ei[NE + e];
        float4 v = reinterpret_cast<const float4*>(ea)[e];
        e0 = v.x; e1 = v.y; e2 = v.z; e3 = v.w;
    } else {
        src = dst = e - NE;
        e0 = eam[0]; e1 = eam[1]; e2 = eam[2]; e3 = eam[3];
    }
    float4 xv = reinterpret_cast<const float4*>(xl + (size_t)src * HH)[lane];
    float4 rv = reinterpret_cast<const float4*>(xr + (size_t)dst * HH)[lane];
    float xs[4] = {xv.x, xv.y, xv.z, xv.w};
    float rs[4] = {rv.x, rv.y, rv.z, rv.w};

    float p = 0.f;
#pragma unroll
    for (int j = 0; j < 4; ++j) {
        int k = lane * 4 + j;
        float m = xs[j] + rs[j]
                + e0 * sWe[k] + e1 * sWe[HH + k]
                + e2 * sWe[2 * HH + k] + e3 * sWe[3 * HH + k];
        m = (m > 0.f) ? m : NEG_SLOPE * m;
        p += m * sAtt[k];
    }
#pragma unroll
    for (int off = 16; off > 0; off >>= 1) p += __shfl_xor(p, off, 32);
    if (lane == 0) {
        sbuf[e] = p;
        atomicMax(&nmax[dst], enc_f(p));
    }
}

// ---------------------------------------------------------------------------
// 6) exp(score - max[dst]); accumulate segment sum
// ---------------------------------------------------------------------------
__global__ void k_edge_exp(const int* __restrict__ ei, float* __restrict__ sbuf,
                           const unsigned* __restrict__ nmax, float* __restrict__ nsum) {
    int e = blockIdx.x * 256 + threadIdx.x;
    if (e >= ESL) return;
    int dst = (e < NE) ? ei[NE + e] : (e - NE);
    float v = __expf(sbuf[e] - dec_f(nmax[dst]));
    sbuf[e] = v;
    atomicAdd(&nsum[dst], v);
}

// ---------------------------------------------------------------------------
// 7) alpha-weighted scatter: out[dst] += alpha * xl[src]
// ---------------------------------------------------------------------------
__global__ void __launch_bounds__(256)
k_edge_scatter(const float* __restrict__ xl, const int* __restrict__ ei,
               const float* __restrict__ sbuf, const float* __restrict__ nsum,
               float* __restrict__ accum) {
    int e = (blockIdx.x * 256 + threadIdx.x) >> 5;
    int lane = threadIdx.x & 31;
    if (e >= ESL) return;
    int src, dst;
    if (e < NE) { src = ei[e]; dst = ei[NE + e]; } else { src = dst = e - NE; }
    float alpha = sbuf[e] / (nsum[dst] + 1e-16f);
    float4 xv = reinterpret_cast<const float4*>(xl + (size_t)src * HH)[lane];
    float* out = accum + (size_t)dst * HH + lane * 4;
    atomicAdd(out + 0, xv.x * alpha);
    atomicAdd(out + 1, xv.y * alpha);
    atomicAdd(out + 2, xv.z * alpha);
    atomicAdd(out + 3, xv.w * alpha);
}

// ---------------------------------------------------------------------------
// 8) h = relu(accum + bias) -> bf16 for next contraction
// ---------------------------------------------------------------------------
__global__ void k_node_update(const float* __restrict__ accum,
                              const float* __restrict__ bias,
                              unsigned short* __restrict__ hb) {
    int idx = blockIdx.x * 256 + threadIdx.x;
    if (idx >= NN * HH) return;
    float v = accum[idx] + bias[idx & 127];
    hb[idx] = f2bf(fmaxf(v, 0.f));
}

// ---------------------------------------------------------------------------
// 9) edge MLP: relu([h[src]|h[dst]|ea] @ Ws1 + bs1) @ Ws2 + bs2
//    16 edges per block; K padded 260 -> 288 (9 k-tiles); 8 waves = 8 N-tiles.
// ---------------------------------------------------------------------------
__global__ void __launch_bounds__(256)
k_edge_mlp(const unsigned short* __restrict__ hb,
           const unsigned short* __restrict__ BpS,
           const float* __restrict__ bs1, const float* __restrict__ Ws2,
           const float* __restrict__ bs2, const int* __restrict__ ei,
           const float* __restrict__ ea, float* __restrict__ out) {
    __shared__ float red[16];
    int lane = threadIdx.x & 31;
    int wv   = threadIdx.x >> 5;
    int kh   = lane >> 4;
    int e    = blockIdx.x * 16 + (lane & 15);
    int src = ei[e], dst = ei[NE + e];
    const unsigned* hsrc = reinterpret_cast<const unsigned*>(hb + (size_t)src * HH);
    const unsigned* hdst = reinterpret_cast<const unsigned*>(hb + (size_t)dst * HH);

    v8f acc = {};
#pragma unroll
    for (int kt = 0; kt < 9; ++kt) {
        FragBF a;
#pragma unroll
        for (int t = 0; t < 8; ++t) {
            int k = kt * 32 + ((t & 4) ? 16 : 0) + (t & 3) * 2 + kh * 8;
            unsigned val;
            if (k < 128)       val = hsrc[k >> 1];
            else if (k < 256)  val = hdst[(k - 128) >> 1];
            else if (k < 260)  val = (unsigned)f2bf(ea[(size_t)e * 4 + (k - 256)])
                                   | ((unsigned)f2bf(ea[(size_t)e * 4 + (k - 255)]) << 16);
            else               val = 0u;
            a.d[t] = val;
        }
        const unsigned* bp = reinterpret_cast<const unsigned*>(
            BpS + (((size_t)wv * 9 + kt) * 32 + lane) * 16);
        FragBF b;
#pragma unroll
        for (int t = 0; t < 8; ++t) b.d[t] = bp[t];
        acc = __builtin_amdgcn_wmma_f32_16x16x32_bf16(false, a.v, false, b.v,
                                                      (short)0, acc, false, false);
    }

    int col = wv * 16 + (lane & 15);
    float w2 = Ws2[col];
    float b1 = bs1[col];
    float s[8];
#pragma unroll
    for (int r = 0; r < 8; ++r) s[r] = fmaxf(acc[r] + b1, 0.f) * w2;
    // reduce over the 16 lanes in each half-wave (stays within half: xor < 16)
#pragma unroll
    for (int off = 1; off < 16; off <<= 1)
#pragma unroll
        for (int r = 0; r < 8; ++r) s[r] += __shfl_xor(s[r], off, 32);

    if (threadIdx.x < 16) red[threadIdx.x] = 0.f;
    __syncthreads();
    if ((lane & 15) == 0) {
#pragma unroll
        for (int r = 0; r < 8; ++r) atomicAdd(&red[kh * 8 + r], s[r]);
    }
    __syncthreads();
    if (threadIdx.x < 16)
        out[blockIdx.x * 16 + threadIdx.x] = red[threadIdx.x] + bs2[0];
}

// ---------------------------------------------------------------------------
// host launcher
// ---------------------------------------------------------------------------
static void* ws_take(void* base, size_t& off, size_t bytes) {
    off = (off + 255) & ~(size_t)255;
    void* p = (char*)base + off;
    off += bytes;
    return p;
}

extern "C" void kernel_launch(void* const* d_in, const int* in_sizes, int n_in,
                              void* d_out, int out_size, void* d_ws, size_t ws_size,
                              hipStream_t stream) {
    (void)in_sizes; (void)n_in; (void)out_size; (void)ws_size;
    const float* x      = (const float*)d_in[0];
    const float* ea     = (const float*)d_in[1];
    const float* Wj     = (const float*)d_in[2];
    const float* bj     = (const float*)d_in[3];
    const float* Wrt    = (const float*)d_in[4];
    const float* brt    = (const float*)d_in[5];
    const float* cWl[2] = {(const float*)d_in[6],  (const float*)d_in[11]};
    const float* cWr[2] = {(const float*)d_in[7],  (const float*)d_in[12]};
    const float* cWe[2] = {(const float*)d_in[8],  (const float*)d_in[13]};
    const float* cAt[2] = {(const float*)d_in[9],  (const float*)d_in[14]};
    const float* cB[2]  = {(const float*)d_in[10], (const float*)d_in[15]};
    const float* Ws1    = (const float*)d_in[16];
    const float* bs1    = (const float*)d_in[17];
    const float* Ws2    = (const float*)d_in[18];
    const float* bs2    = (const float*)d_in[19];
    const int*   ei     = (const int*)d_in[20];
    float*       out    = (float*)d_out;

    size_t off = 0;
    unsigned short* hb   = (unsigned short*)ws_take(d_ws, off, (size_t)NN * HH * 2);
    float*  xl           = (float*)ws_take(d_ws, off, (size_t)NN * HH * 4);
    float*  xr           = (float*)ws_take(d_ws, off, (size_t)NN * HH * 4);
    float*  accum        = (float*)ws_take(d_ws, off, (size_t)NN * HH * 4);
    float*  sbuf         = (float*)ws_take(d_ws, off, (size_t)ESL * 4);
    unsigned* nmax       = (unsigned*)ws_take(d_ws, off, (size_t)NN * 4);
    float*  nsum         = (float*)ws_take(d_ws, off, (size_t)NN * 4);
    float*  eam          = (float*)ws_take(d_ws, off, 4 * 4);
    unsigned short* BpL[2], *BpR[2];
    BpL[0] = (unsigned short*)ws_take(d_ws, off, (size_t)HH * HH * 2);
    BpR[0] = (unsigned short*)ws_take(d_ws, off, (size_t)HH * HH * 2);
    BpL[1] = (unsigned short*)ws_take(d_ws, off, (size_t)HH * HH * 2);
    BpR[1] = (unsigned short*)ws_take(d_ws, off, (size_t)HH * HH * 2);
    unsigned short* BpS = (unsigned short*)ws_take(d_ws, off, (size_t)288 * HH * 2);

    const int nodeBlocks = (NN * HH) / 256;        // 25000
    const int mTiles     = NN / 16;                // 3125
    const int waveBlocks = ESL / 8;                // 106250 (exact)
    const int edgeBlocks = (ESL + 255) / 256;      // 3321
    const int mlpBlocks  = NE / 16;                // 50000

    k_ea_mean<<<1, 256, 0, stream>>>(ea, eam);
    k_encode<<<nodeBlocks, 256, 0, stream>>>(x, Wj, bj, Wrt, brt, hb);

    k_pack_b<<<64, 256, 0, stream>>>(cWl[0], BpL[0], HH, 4);
    k_pack_b<<<64, 256, 0, stream>>>(cWr[0], BpR[0], HH, 4);
    k_pack_b<<<64, 256, 0, stream>>>(cWl[1], BpL[1], HH, 4);
    k_pack_b<<<64, 256, 0, stream>>>(cWr[1], BpR[1], HH, 4);
    k_pack_b<<<144, 256, 0, stream>>>(Ws1, BpS, 2 * HH + EDIM, 9);

    for (int L = 0; L < 2; ++L) {
        k_layer_init<<<nodeBlocks, 256, 0, stream>>>(nmax, nsum, accum);
        k_gemm_xlxr<<<mTiles, 256, 0, stream>>>(hb, BpL[L], BpR[L], xl, xr);
        k_edge_score<<<waveBlocks, 256, 0, stream>>>(xl, xr, ea, ei, cWe[L], cAt[L],
                                                     eam, sbuf, nmax);
        k_edge_exp<<<edgeBlocks, 256, 0, stream>>>(ei, sbuf, nmax, nsum);
        k_edge_scatter<<<waveBlocks, 256, 0, stream>>>(xl, ei, sbuf, nsum, accum);
        k_node_update<<<nodeBlocks, 256, 0, stream>>>(accum, cB[L], hb);
    }

    k_edge_mlp<<<mlpBlocks, 256, 0, stream>>>(hb, BpS, bs1, Ws2, bs2, ei, ea, out);
}
